// FlowReprLogitAggregator_89111981457417
// MI455X (gfx1250) — matrix operand order repaired and checked by reading, not compiled, not run on hardware
//
#include <hip/hip_runtime.h>
#include <hip/hip_bf16.h>
#include <math.h>
#include <stdint.h>

// ---------------- Problem constants (from reference) ----------------
#define N_PKT   500000
#define D_DIM   256
#define C_DIM   64
#define H_DIM   128
#define F_FLOWS 50000
#define EPS_LN  1e-5f
#define EPS_DEN 1e-12f

typedef __attribute__((ext_vector_type(16))) __bf16 v16bf;
typedef __attribute__((ext_vector_type(8)))  float  v8f;

// ---------------- helpers ----------------
__device__ __forceinline__ unsigned keyFromFloat(float f) {
    unsigned b = __float_as_uint(f);
    return (b & 0x80000000u) ? ~b : (b | 0x80000000u);
}
__device__ __forceinline__ float floatFromKey(unsigned k) {
    unsigned b = (k & 0x80000000u) ? (k ^ 0x80000000u) : ~k;
    return __uint_as_float(b);
}

// One 16x16 output tile accumulated over NKT k-steps of v_wmma_f32_16x16x32_bf16.
// a_row: per-lane pointer to row (M = lane&15) of the bf16 A panel in LDS.
// bpack: weights pre-packed in B-fragment order: [(ks*nnt+nt)*32+lane]*16 bf16 contiguous.
template <int NKT>
__device__ __forceinline__ v8f gemm_nt(const __bf16* a_row, const unsigned short* bpack,
                                       int nnt, int nt, int lane, float bias) {
    v8f acc;
#pragma unroll
    for (int v = 0; v < 8; ++v) acc[v] = bias;
#pragma unroll
    for (int ks = 0; ks < NKT; ++ks) {
        const int K0 = ks * 32 + ((lane & 16) ? 8 : 0);
        union { uint4 u[2]; v16bf v; } A, B;
        A.u[0] = *(const uint4*)(a_row + K0);
        A.u[1] = *(const uint4*)(a_row + K0 + 16);
        const uint4* bp = (const uint4*)bpack + ((size_t)(ks * nnt + nt) * 32 + lane) * 2;
        B.u[0] = bp[0];
        B.u[1] = bp[1];
        acc = __builtin_amdgcn_wmma_f32_16x16x32_bf16(false, A.v, false, B.v,
                                                      (short)0, acc, false, false);
    }
    return acc;
}

// ---------------- weight pre-pack: fp32 row-major [K x Nc] -> bf16 B-fragments ----------------
__global__ void pack_weights_kernel(const float* __restrict__ W, unsigned short* __restrict__ out,
                                    int K, int Nc) {
    const int total = K * Nc;
    const int nnt = Nc >> 4;
    for (int i = blockIdx.x * blockDim.x + threadIdx.x; i < total; i += gridDim.x * blockDim.x) {
        const int e    = i & 15;
        const int lane = (i >> 4) & 31;
        const int fr   = i >> 9;               // (ks*nnt + nt)
        const int nt   = fr % nnt;
        const int ks   = fr / nnt;
        const int k = ks * 32 + ((lane & 16) ? 8 : 0) + (e & 7) + ((e & 8) ? 16 : 0);   // B frag: table order (lanes16-31 carry K+8..15, K+24..31)
        const int n = nt * 16 + (lane & 15);
        const __bf16 b = (__bf16)W[(size_t)k * Nc + n];
        out[i] = __builtin_bit_cast(unsigned short, b);
    }
}

// ---------------- zero / init accumulators ----------------
__global__ void init_kernel(float* mean_acc, float* attn, float* logit_acc,
                            float* denom, float* counts, unsigned* maxkey) {
    const int i = blockIdx.x * blockDim.x + threadIdx.x;
    if (i < F_FLOWS * 128) { mean_acc[i] = 0.f; attn[i] = 0.f; }
    if (i < F_FLOWS * 64)  logit_acc[i] = 0.f;
    if (i < F_FLOWS)       { denom[i] = 0.f; counts[i] = 0.f; maxkey[i] = 0u; }
}

// ---------------- per-packet fused kernel (WMMA hot path) ----------------
// 64 threads = 2 waves; each wave owns a 16-packet M-tile.
__global__ __launch_bounds__(64) void packet_kernel(
    const float* __restrict__ pr, const float* __restrict__ logits,
    const int* __restrict__ idx,
    const float* __restrict__ ln_r_g, const float* __restrict__ ln_r_b, const float* __restrict__ b_r,
    const float* __restrict__ ln_l_g, const float* __restrict__ ln_l_b, const float* __restrict__ b_l,
    const float* __restrict__ ln_f_g, const float* __restrict__ ln_f_b, const float* __restrict__ b_f,
    const float* __restrict__ w_a, const float* __restrict__ b_a,
    const unsigned short* __restrict__ wrp, const unsigned short* __restrict__ wlp,
    const unsigned short* __restrict__ wfp,
    unsigned short* __restrict__ h_out, float* __restrict__ score_out,
    unsigned* __restrict__ maxkey, float* __restrict__ counts,
    float* __restrict__ mean_acc, float* __restrict__ logit_acc) {

    __shared__ __align__(16) float  s_hin[2][16][260];   // r|l panel (fp32), padded rows
    __shared__ __align__(16) __bf16 s_a[2][16][264];     // bf16 A panel for WMMA, padded rows

    const int tid = threadIdx.x;
    const int w = tid >> 5, lane = tid & 31;
    const int row = lane & 15, hh = lane >> 4;           // A-row owned by lane; half index
    const int m0 = (blockIdx.x * 2 + w) * 16;            // N_PKT % 32 == 0
    const int m = m0 + row;

    // ---- phase 1: LN(packet_repr) -> bf16 A panel ----
    {
        const float* p = pr + (size_t)m * D_DIM + hh * 128;
        float sum = 0.f, sq = 0.f;
        for (int j = 0; j < 128; j += 4) {
            float4 v = *(const float4*)(p + j);
            *(float4*)&s_hin[w][row][hh * 128 + j] = v;  // stash raw for 2nd LN pass
            sum += v.x + v.y + v.z + v.w;
            sq  += v.x * v.x + v.y * v.y + v.z * v.z + v.w * v.w;
        }
        sum += __shfl_xor(sum, 16); sq += __shfl_xor(sq, 16);
        const float mean = sum * (1.f / 256.f);
        const float var  = sq * (1.f / 256.f) - mean * mean;
        const float rstd = rsqrtf(var + EPS_LN);
        for (int j = 0; j < 128; ++j) {
            const int c = hh * 128 + j;
            const float y = (s_hin[w][row][c] - mean) * rstd * ln_r_g[c] + ln_r_b[c];
            s_a[w][row][c] = (__bf16)y;
        }
    }
    __syncthreads();

    const __bf16* a_row = &s_a[w][row][0];

    // ---- GEMM r = LN(x) @ W_r + b_r  -> s_hin[:, 0:128] ----
    for (int nt = 0; nt < 8; ++nt) {
        v8f acc = gemm_nt<8>(a_row, wrp, 8, nt, lane, b_r[nt * 16 + row]);
#pragma unroll
        for (int v = 0; v < 8; ++v)
            s_hin[w][v + hh * 8][nt * 16 + row] = acc[v];
    }
    __syncthreads();

    // ---- phase 2: softmax(logits), LN, logit/count accumulation ----
    {
        const float* lg = logits + (size_t)m * C_DIM + hh * 32;
        const int flow = idx[m];
        float lv[32];
        float mx = -3.4e38f;
        for (int j = 0; j < 32; ++j) { lv[j] = lg[j]; mx = fmaxf(mx, lv[j]); }
        mx = fmaxf(mx, __shfl_xor(mx, 16));
        float se = 0.f;
        for (int j = 0; j < 32; ++j) { lv[j] = __expf(lv[j] - mx); se += lv[j]; }
        se += __shfl_xor(se, 16);
        const float inv = 1.f / se;
        float sq = 0.f;
        for (int j = 0; j < 32; ++j) { lv[j] *= inv; sq += lv[j] * lv[j]; }
        sq += __shfl_xor(sq, 16);
        const float mean = 1.f / 64.f;
        const float var  = sq * (1.f / 64.f) - mean * mean;
        const float rstd = rsqrtf(var + EPS_LN);
        for (int j = 0; j < 32; ++j) {
            const int c = hh * 32 + j;
            s_a[w][row][c] = (__bf16)((lv[j] - mean) * rstd * ln_l_g[c] + ln_l_b[c]);
            atomicAdd(&logit_acc[(size_t)flow * 64 + c], lg[j]);  // raw logits mean numerator
        }
        if (hh == 0) atomicAdd(&counts[flow], 1.f);
    }
    __syncthreads();

    // ---- GEMM l = LN(probs) @ W_l + b_l -> s_hin[:, 128:256] ----
    for (int nt = 0; nt < 8; ++nt) {
        v8f acc = gemm_nt<2>(a_row, wlp, 8, nt, lane, b_l[nt * 16 + row]);
#pragma unroll
        for (int v = 0; v < 8; ++v)
            s_hin[w][v + hh * 8][128 + nt * 16 + row] = acc[v];
    }
    __syncthreads();

    // ---- phase 3: LN over concat(r,l) -> bf16 A panel ----
    {
        float sum = 0.f, sq = 0.f;
        const float* hb = &s_hin[w][row][hh * 128];
        for (int j = 0; j < 128; j += 4) {
            float4 v = *(const float4*)(hb + j);
            sum += v.x + v.y + v.z + v.w;
            sq  += v.x * v.x + v.y * v.y + v.z * v.z + v.w * v.w;
        }
        sum += __shfl_xor(sum, 16); sq += __shfl_xor(sq, 16);
        const float mean = sum * (1.f / 256.f);
        const float var  = sq * (1.f / 256.f) - mean * mean;
        const float rstd = rsqrtf(var + EPS_LN);
        for (int j = 0; j < 128; ++j) {
            const int c = hh * 128 + j;
            s_a[w][row][c] = (__bf16)((s_hin[w][row][c] - mean) * rstd * ln_f_g[c] + ln_f_b[c]);
        }
    }
    __syncthreads();

    // ---- GEMM h = relu(LN @ W_f + b_f); score = h.w_a + b_a; stream h (bf16) ----
    {
        const float ba = b_a[0];
        float c8[8];
#pragma unroll
        for (int v = 0; v < 8; ++v) c8[v] = 0.f;
        for (int nt = 0; nt < 8; ++nt) {
            v8f acc = gemm_nt<8>(a_row, wfp, 8, nt, lane, b_f[nt * 16 + row]);
            const float wv = w_a[nt * 16 + row];
#pragma unroll
            for (int v = 0; v < 8; ++v) {
                const float hx = fmaxf(acc[v], 0.f);
                c8[v] += hx * wv;
                h_out[(size_t)(m0 + v + hh * 8) * H_DIM + nt * 16 + row] =
                    __builtin_bit_cast(unsigned short, (__bf16)hx);
            }
        }
        // reduce columns across the 16 lanes of each half
#pragma unroll
        for (int off = 1; off < 16; off <<= 1)
#pragma unroll
            for (int v = 0; v < 8; ++v) c8[v] += __shfl_xor(c8[v], off);
        if (row < 8) {
            const int M = hh * 8 + row;
            const float sc = c8[row] + ba;
            score_out[m0 + M] = sc;
            atomicMax(&maxkey[idx[m0 + M]], keyFromFloat(sc));
        }
    }

    // ---- accumulate r into per-flow mean numerator ----
    for (int t = lane; t < 16 * 128; t += 32) {
        const int rr = t >> 7, c = t & 127;
        atomicAdd(&mean_acc[(size_t)idx[m0 + rr] * 128 + c], s_hin[w][rr][c]);
    }
}

// ---------------- segment softmax denominator ----------------
__global__ void denom_kernel(const float* __restrict__ score, const int* __restrict__ idx,
                             const unsigned* __restrict__ maxkey,
                             float* __restrict__ expv, float* __restrict__ denom) {
    const int p = blockIdx.x * blockDim.x + threadIdx.x;
    if (p >= N_PKT) return;
    const int f = idx[p];
    const float e = __expf(score[p] - floatFromKey(maxkey[f]));
    expv[p] = e;
    atomicAdd(&denom[f], e);
}

// ---------------- alpha-weighted h accumulation ----------------
__global__ void attn_kernel(const unsigned short* __restrict__ h, const int* __restrict__ idx,
                            const float* __restrict__ expv, const float* __restrict__ denom,
                            float* __restrict__ attn) {
    const long long t = (long long)blockIdx.x * blockDim.x + threadIdx.x;
    if (t >= (long long)N_PKT * H_DIM) return;
    const int p = (int)(t >> 7), c = (int)(t & 127);
    const int f = idx[p];
    const float alpha = expv[p] / (denom[f] + EPS_DEN);
    const __bf16 hb = __builtin_bit_cast(__bf16, h[t]);
    atomicAdd(&attn[(size_t)f * H_DIM + c], (float)hb * alpha);
}

// ---------------- flow-head MLP (WMMA) + final residual ----------------
__global__ __launch_bounds__(64) void flow_head_kernel(
    const float* __restrict__ mean_acc, const float* __restrict__ attn,
    const float* __restrict__ logit_acc, const float* __restrict__ counts,
    const float* __restrict__ ln_h_g, const float* __restrict__ ln_h_b,
    const float* __restrict__ b_h1, const float* __restrict__ b_h2,
    const float* __restrict__ gamma_res_p, const float* __restrict__ gamma_logit_p,
    const unsigned short* __restrict__ wh1p, const unsigned short* __restrict__ wh2p,
    float* __restrict__ out) {

    __shared__ __align__(16) __bf16 s_a[2][16][136];
    __shared__ __align__(16) __bf16 s_fh[2][16][136];

    const int tid = threadIdx.x;
    const int w = tid >> 5, lane = tid & 31;
    const int row = lane & 15, hh = lane >> 4;
    const int f0 = (blockIdx.x * 2 + w) * 16;
    const int f  = f0 + row;
    const int fr = (f < F_FLOWS) ? f : (F_FLOWS - 1);   // clamp: EXEC stays all-ones for WMMA
    const float gres = gamma_res_p[0], glog = gamma_logit_p[0];
    const float inv = 1.f / fmaxf(counts[fr], 1.f);

    // flow_repr + LN(H=128); each lane owns 64 columns of its row
    float xv[64];
    float sum = 0.f, sq = 0.f;
    for (int j = 0; j < 64; ++j) {
        const int c = hh * 64 + j;
        const float x = mean_acc[(size_t)fr * 128 + c] * inv + gres * attn[(size_t)fr * 128 + c];
        xv[j] = x; sum += x; sq += x * x;
    }
    sum += __shfl_xor(sum, 16); sq += __shfl_xor(sq, 16);
    const float mean = sum * (1.f / 128.f);
    const float var  = sq * (1.f / 128.f) - mean * mean;
    const float rstd = rsqrtf(var + EPS_LN);
    for (int j = 0; j < 64; ++j) {
        const int c = hh * 64 + j;
        s_a[w][row][c] = (__bf16)((xv[j] - mean) * rstd * ln_h_g[c] + ln_h_b[c]);
    }
    __syncthreads();

    const __bf16* a_row = &s_a[w][row][0];
    for (int nt = 0; nt < 8; ++nt) {
        v8f acc = gemm_nt<4>(a_row, wh1p, 8, nt, lane, b_h1[nt * 16 + row]);
#pragma unroll
        for (int v = 0; v < 8; ++v)
            s_fh[w][v + hh * 8][nt * 16 + row] = (__bf16)fmaxf(acc[v], 0.f);
    }
    __syncthreads();

    const __bf16* a2 = &s_fh[w][row][0];
    for (int nt = 0; nt < 4; ++nt) {
        v8f acc = gemm_nt<4>(a2, wh2p, 4, nt, lane, b_h2[nt * 16 + row]);
#pragma unroll
        for (int v = 0; v < 8; ++v) {
            const int fv = f0 + v + hh * 8;
            if (fv < F_FLOWS) {
                const float cv = fmaxf(counts[fv], 1.f);
                const int c = nt * 16 + row;
                out[(size_t)fv * 64 + c] = logit_acc[(size_t)fv * 64 + c] / cv + glog * acc[v];
            }
        }
    }
}

// ---------------- host-side launch ----------------
extern "C" void kernel_launch(void* const* d_in, const int* in_sizes, int n_in,
                              void* d_out, int out_size, void* d_ws, size_t ws_size,
                              hipStream_t stream) {
    const float* pr      = (const float*)d_in[0];
    const float* logits  = (const float*)d_in[1];
    const int*   idx     = (const int*)d_in[2];
    const float* ln_r_g  = (const float*)d_in[4];
    const float* ln_r_b  = (const float*)d_in[5];
    const float* W_r     = (const float*)d_in[6];
    const float* b_r     = (const float*)d_in[7];
    const float* ln_l_g  = (const float*)d_in[8];
    const float* ln_l_b  = (const float*)d_in[9];
    const float* W_l     = (const float*)d_in[10];
    const float* b_l     = (const float*)d_in[11];
    const float* ln_f_g  = (const float*)d_in[12];
    const float* ln_f_b  = (const float*)d_in[13];
    const float* W_f     = (const float*)d_in[14];
    const float* b_f     = (const float*)d_in[15];
    const float* w_a     = (const float*)d_in[16];
    const float* b_a     = (const float*)d_in[17];
    const float* g_res   = (const float*)d_in[18];
    const float* ln_h_g  = (const float*)d_in[19];
    const float* ln_h_b  = (const float*)d_in[20];
    const float* W_h1    = (const float*)d_in[21];
    const float* b_h1    = (const float*)d_in[22];
    const float* W_h2    = (const float*)d_in[23];
    const float* b_h2    = (const float*)d_in[24];
    const float* g_log   = (const float*)d_in[25];
    float* out = (float*)d_out;

    char* ws = (char*)d_ws;
    size_t off = 0;
    auto take = [&](size_t bytes) -> char* {
        char* p = ws + off;
        off = (off + bytes + 255) & ~(size_t)255;
        return p;
    };
    unsigned short* wrp  = (unsigned short*)take((size_t)D_DIM * H_DIM * 2);
    unsigned short* wlp  = (unsigned short*)take((size_t)C_DIM * H_DIM * 2);
    unsigned short* wfp  = (unsigned short*)take((size_t)2 * H_DIM * H_DIM * 2);
    unsigned short* wh1p = (unsigned short*)take((size_t)H_DIM * H_DIM * 2);
    unsigned short* wh2p = (unsigned short*)take((size_t)H_DIM * C_DIM * 2);
    unsigned* maxkey     = (unsigned*)take((size_t)F_FLOWS * 4);
    float* denom         = (float*)take((size_t)F_FLOWS * 4);
    float* counts        = (float*)take((size_t)F_FLOWS * 4);
    float* score         = (float*)take((size_t)N_PKT * 4);
    float* expv          = (float*)take((size_t)N_PKT * 4);
    float* mean_acc      = (float*)take((size_t)F_FLOWS * 128 * 4);
    float* attn          = (float*)take((size_t)F_FLOWS * 128 * 4);
    float* logit_acc     = (float*)take((size_t)F_FLOWS * 64 * 4);
    unsigned short* h_ws = (unsigned short*)take((size_t)N_PKT * H_DIM * 2);

    // 1) pack weights to WMMA-B fragment order (bf16)
    pack_weights_kernel<<<128, 256, 0, stream>>>(W_r,  wrp,  D_DIM, H_DIM);
    pack_weights_kernel<<<32,  256, 0, stream>>>(W_l,  wlp,  C_DIM, H_DIM);
    pack_weights_kernel<<<128, 256, 0, stream>>>(W_f,  wfp,  2 * H_DIM, H_DIM);
    pack_weights_kernel<<<64,  256, 0, stream>>>(W_h1, wh1p, H_DIM, H_DIM);
    pack_weights_kernel<<<32,  256, 0, stream>>>(W_h2, wh2p, H_DIM, C_DIM);

    // 2) zero/init accumulators
    init_kernel<<<(F_FLOWS * 128 + 255) / 256, 256, 0, stream>>>(
        mean_acc, attn, logit_acc, denom, counts, maxkey);

    // 3) fused per-packet WMMA pipeline (32 packets / block)
    packet_kernel<<<N_PKT / 32, 64, 0, stream>>>(
        pr, logits, idx, ln_r_g, ln_r_b, b_r, ln_l_g, ln_l_b, b_l,
        ln_f_g, ln_f_b, b_f, w_a, b_a, wrp, wlp, wfp,
        h_ws, score, maxkey, counts, mean_acc, logit_acc);

    // 4) segment softmax denominator
    denom_kernel<<<(N_PKT + 255) / 256, 256, 0, stream>>>(score, idx, maxkey, expv, denom);

    // 5) alpha-weighted h accumulation
    attn_kernel<<<(int)(((long long)N_PKT * H_DIM + 255) / 256), 256, 0, stream>>>(
        h_ws, idx, expv, denom, attn);

    // 6) flow-head MLP + residuals (32 flows / block)
    flow_head_kernel<<<(F_FLOWS + 31) / 32, 64, 0, stream>>>(
        mean_acc, attn, logit_acc, counts, ln_h_g, ln_h_b, b_h1, b_h2,
        g_res, g_log, wh1p, wh2p, out);
}
